// Decoder_5480378270296
// MI455X (gfx1250) — matrix operand order repaired and checked by reading, not compiled, not run on hardware
//
#include <hip/hip_runtime.h>
#include <hip/hip_bf16.h>

// ---------------- problem constants ----------------
#define MSL   64
#define BS    128
#define ENCD  1024
#define HID   512
#define EMBD  50
#define ATOM  64
#define NSTEP 49            // step 49 (0-based) of the reference is discarded -> skip it
#define BTOT  (MSL*BS)      // 8192 independent rows

// ---------------- vector types ----------------
typedef __attribute__((ext_vector_type(16))) __bf16        bf16x16;
typedef __attribute__((ext_vector_type(8)))  float         f32x8;
typedef __attribute__((ext_vector_type(4)))  float         f32x4;
typedef __attribute__((ext_vector_type(4)))  unsigned int  u32x4;

union Frag { u32x4 u[2]; bf16x16 v; };

// ---------------- explicit global-address-space accessors ----------------
#define GAS __attribute__((address_space(1)))
__device__ __forceinline__ u32x4 ldg_u4(const void* p) {
  return *(const GAS u32x4*)(unsigned long long)p;
}
__device__ __forceinline__ float ldg_f(const void* p) {
  return *(const GAS float*)(unsigned long long)p;
}
__device__ __forceinline__ void stg_f(void* p, float v) {
  *(GAS float*)(unsigned long long)p = v;
}
__device__ __forceinline__ void stg_bf(void* p, __bf16 v) {
  *(GAS __bf16*)(unsigned long long)p = v;
}
__device__ __forceinline__ void stg_u4(void* p, u32x4 v) {
  *(GAS u32x4*)(unsigned long long)p = v;
}
__device__ __forceinline__ void stg_f4(void* p, f32x4 v) {
  *(GAS f32x4*)(unsigned long long)p = v;
}

// ---------------- workspace layout (bytes), ~92 MB total ----------------
#define OFF_ENCB   ((size_t)0)
#define OFF_WH0T   (OFF_ENCB  + (size_t)BTOT*ENCD*2)
#define OFF_WIH0T  (OFF_WH0T  + (size_t)ENCD*(2*HID)*2)
#define OFF_WHH0T  (OFF_WIH0T + (size_t)64*(3*HID)*2)
#define OFF_WIH1T  (OFF_WHH0T + (size_t)HID*(3*HID)*2)
#define OFF_WHH1T  (OFF_WIH1T + (size_t)HID*(3*HID)*2)
#define OFF_WOUTT  (OFF_WHH1T + (size_t)HID*(3*HID)*2)
#define OFF_EMBB   (OFF_WOUTT + (size_t)HID*ATOM*2)
#define OFF_H0F    (OFF_EMBB  + (size_t)ATOM*64*2)
#define OFF_H1F    (OFF_H0F   + (size_t)BTOT*HID*4)
#define OFF_H0BA   (OFF_H1F   + (size_t)BTOT*HID*4)
#define OFF_H0BB   (OFF_H0BA  + (size_t)BTOT*HID*2)
#define OFF_H1BA   (OFF_H0BB  + (size_t)BTOT*HID*2)
#define OFF_H1BB   (OFF_H1BA  + (size_t)BTOT*HID*2)
#define OFF_EBF    (OFF_H1BB  + (size_t)BTOT*HID*2)

// ---------------- WMMA ----------------
__device__ __forceinline__ f32x8 wmma_bf16(bf16x16 a, bf16x16 b, f32x8 c) {
  return __builtin_amdgcn_wmma_f32_16x16x32_bf16(false, a, false, b, (short)0, c, false, false);
}

__device__ __forceinline__ float sigmoidf_(float x) { return 1.0f / (1.0f + __expf(-x)); }

// ------------- fragment bundles, loaded via walked byte pointers -------------
// pa: per-lane A base = x + ((m0 + lane%16)*stride + (lane/16)*8)*2 ; advances 64 B per k-step
//     A subtile 1 (+16 rows) at constant byte offset AX2 = 16*stride*2
// pb: per-lane B base = wt + (lane*(3*HID) + n0)*2 ; advances 32*(3*HID)*2 B per k-step
//     z/n gate columns at constant byte offsets +HID*2 / +2*HID*2
struct Frag5 { Frag br, bz, bn, a0, a1; };
struct Frag3 { Frag b, a0, a1; };

template<int AX2>
__device__ __forceinline__ void ld5p(Frag5& f, const char* pa, const char* pb) {
  f.br.u[0] = ldg_u4(pb);               f.br.u[1] = ldg_u4(pb + 16);
  f.bz.u[0] = ldg_u4(pb + 2 * HID);     f.bz.u[1] = ldg_u4(pb + 2 * HID + 16);
  f.bn.u[0] = ldg_u4(pb + 4 * HID);     f.bn.u[1] = ldg_u4(pb + 4 * HID + 16);
  f.a0.u[0] = ldg_u4(pa);               f.a0.u[1] = ldg_u4(pa + 32);
  f.a1.u[0] = ldg_u4(pa + AX2);         f.a1.u[1] = ldg_u4(pa + AX2 + 32);
}
template<int AX2>
__device__ __forceinline__ void ld3p(Frag3& f, const char* pa, const char* pb) {
  f.b.u[0]  = ldg_u4(pb);               f.b.u[1]  = ldg_u4(pb + 16);
  f.a0.u[0] = ldg_u4(pa);               f.a0.u[1] = ldg_u4(pa + 32);
  f.a1.u[0] = ldg_u4(pa + AX2);         f.a1.u[1] = ldg_u4(pa + AX2 + 32);
}

__device__ __forceinline__ void fma6(const Frag5& f, f32x8* ar, f32x8* az, f32x8* an) {
  ar[0] = wmma_bf16(f.a0.v, f.br.v, ar[0]);
  ar[1] = wmma_bf16(f.a1.v, f.br.v, ar[1]);
  az[0] = wmma_bf16(f.a0.v, f.bz.v, az[0]);
  az[1] = wmma_bf16(f.a1.v, f.bz.v, az[1]);
  an[0] = wmma_bf16(f.a0.v, f.bn.v, an[0]);
  an[1] = wmma_bf16(f.a1.v, f.bn.v, an[1]);
}
__device__ __forceinline__ void fma2(const Frag3& f, f32x8* acc) {
  acc[0] = wmma_bf16(f.a0.v, f.b.v, acc[0]);
  acc[1] = wmma_bf16(f.a1.v, f.b.v, acc[1]);
}

// Ping-pong pipelined GEMM phase, NK k-steps (NK even >= 2): f1's loads are in
// flight while f0's WMMAs issue -> partial s_wait_loadcnt, zero register copies.
#define PB5STEP (32 * 3 * HID * 2)
template<int NK, int AX2>
__device__ __forceinline__ void gemm5(const char* pa, const char* pb,
                                      f32x8* ar, f32x8* az, f32x8* an) {
  Frag5 f0, f1;
  ld5p<AX2>(f0, pa, pb);
  #pragma unroll 1
  for (int i = 0; i < NK / 2 - 1; ++i) {
    pa += 64; pb += PB5STEP;
    ld5p<AX2>(f1, pa, pb);
    fma6(f0, ar, az, an);
    pa += 64; pb += PB5STEP;
    ld5p<AX2>(f0, pa, pb);
    fma6(f1, ar, az, an);
  }
  pa += 64; pb += PB5STEP;
  ld5p<AX2>(f1, pa, pb);
  fma6(f0, ar, az, an);
  fma6(f1, ar, az, an);
}
template<int NK, int AX2, int PBSTEP>
__device__ __forceinline__ void gemm1(const char* pa, const char* pb, f32x8* acc) {
  Frag3 f0, f1;
  ld3p<AX2>(f0, pa, pb);
  #pragma unroll 1
  for (int i = 0; i < NK / 2 - 1; ++i) {
    pa += 64; pb += PBSTEP;
    ld3p<AX2>(f1, pa, pb);
    fma2(f0, acc);
    pa += 64; pb += PBSTEP;
    ld3p<AX2>(f0, pa, pb);
    fma2(f1, acc);
  }
  pa += 64; pb += PBSTEP;
  ld3p<AX2>(f1, pa, pb);
  fma2(f0, acc);
  fma2(f1, acc);
}

// ---------------- prep kernels ----------------
__global__ void k_cvt(const float* __restrict__ src, __bf16* __restrict__ dst, int n) {
  int i = blockIdx.x * blockDim.x + threadIdx.x;
  if (i < n) stg_bf(dst + i, (__bf16)ldg_f(src + i));
}
__global__ void k_cvtT(const float* __restrict__ w, __bf16* __restrict__ wt, int N, int K, int Kpad) {
  int i = blockIdx.x * blockDim.x + threadIdx.x;
  if (i >= Kpad * N) return;
  int r = i / N, c = i - r * N, L = r & 31, j = c & 15;
  int n = (c & ~15) + (L & 15), k = (r & ~31) + (j < 8 ? 8 * (L >> 4) + j : 16 + 8 * (L >> 4) + (j - 8));
  stg_bf(wt + i, (k < K) ? (__bf16)ldg_f(w + (size_t)n * K + k) : (__bf16)0.0f);
}
__global__ void k_emb(const float* __restrict__ emb, __bf16* __restrict__ embb) {
  int i = blockIdx.x * blockDim.x + threadIdx.x;
  if (i >= ATOM * 64) return;
  int a = i >> 6, c = i & 63;
  stg_bf(embb + i, (c < EMBD) ? (__bf16)ldg_f(emb + (size_t)a * EMBD + c) : (__bf16)0.0f);
}

// ---------------- fused GRU layer over this wave's 32 rows ----------------
template<int KX>
__device__ void gru_layer(const __bf16* __restrict__ xb,
                          float* __restrict__ hf,
                          const __bf16* __restrict__ hbin, __bf16* __restrict__ hbout,
                          const __bf16* __restrict__ wih_t, const __bf16* __restrict__ whh_t,
                          const float* __restrict__ b_ih, const float* __restrict__ b_hh,
                          int m0, int lane) {
  const int halfid = lane >> 4, lcol = lane & 15;
  // per-lane bases (n-tile invariant)
  const char* paX  = (const char*)xb   + ((size_t)(m0 + lcol) * KX  + halfid * 8) * 2;
  const char* paH  = (const char*)hbin + ((size_t)(m0 + lcol) * HID + halfid * 8) * 2;
  const char* pbI0 = (const char*)wih_t + (size_t)lane * (3 * HID) * 2;
  const char* pbH0 = (const char*)whh_t + (size_t)lane * (3 * HID) * 2;
  char* hfB = (char*)hf    + (size_t)(m0 + 8 * halfid) * HID * 4;
  char* hbB = (char*)hbout + (size_t)(m0 + 8 * halfid) * HID * 2;

  #pragma unroll 1
  for (int nt = 0; nt < HID / 16; ++nt) {
    const int n0 = nt * 16;
    f32x8 ar[2] = {}, az[2] = {}, ain[2] = {}, ahn[2] = {};
    gemm5<KX / 32, 16 * KX * 2>(paX, pbI0 + n0 * 2, ar, az, ain);   // x @ W_ih^T
    gemm5<HID / 32, 16 * HID * 2>(paH, pbH0 + n0 * 2, ar, az, ahn); // h @ W_hh^T
    // gate epilogue: VGPR i, lane L -> row m0+16s+i+8*(L/16), col n0+(L%16)
    const int col = n0 + lcol;
    const float brz0 = ldg_f(b_ih + col)           + ldg_f(b_hh + col);
    const float bzz0 = ldg_f(b_ih + HID + col)     + ldg_f(b_hh + HID + col);
    const float bin0 = ldg_f(b_ih + 2 * HID + col);
    const float bhn0 = ldg_f(b_hh + 2 * HID + col);
    char* hfp = hfB + (size_t)col * 4;
    char* hbp = hbB + (size_t)col * 2;
    #pragma unroll
    for (int s = 0; s < 2; ++s)
      #pragma unroll
      for (int i = 0; i < 8; ++i) {
        const int ro = 16 * s + i;                       // row offset, compile-time
        const float hold = ldg_f(hfp + ro * (HID * 4));
        const float r = sigmoidf_((s ? ar[1][i] : ar[0][i]) + brz0);
        const float z = sigmoidf_((s ? az[1][i] : az[0][i]) + bzz0);
        const float n = tanhf((s ? ain[1][i] : ain[0][i]) + bin0 +
                              r * ((s ? ahn[1][i] : ahn[0][i]) + bhn0));
        const float hnew = (1.0f - z) * n + z * hold;
        stg_f(hfp + ro * (HID * 4), hnew);
        stg_bf(hbp + ro * (HID * 2), (__bf16)hnew);
      }
  }
}

// ---------------- output stage ----------------
__device__ void out_stage(const __bf16* __restrict__ h1b,
                          const __bf16* __restrict__ woutt, const float* __restrict__ b_out,
                          const __bf16* __restrict__ embb, __bf16* __restrict__ ebf,
                          float* __restrict__ yout, float* __restrict__ lds,
                          int m0, int lane) {
  const int halfid = lane >> 4, lcol = lane & 15;
  const char* paH = (const char*)h1b + ((size_t)(m0 + lcol) * HID + halfid * 8) * 2;
  const char* pbO = (const char*)woutt + (size_t)lane * ATOM * 2;
  #pragma unroll 1
  for (int t = 0; t < ATOM / 16; ++t) {
    const int n0 = t * 16;
    f32x8 acc[2] = {};
    gemm1<HID / 32, 16 * HID * 2, 32 * ATOM * 2>(paH, pbO + n0 * 2, acc);
    const float bo = ldg_f(b_out + n0 + lcol);
    #pragma unroll
    for (int s = 0; s < 2; ++s)
      #pragma unroll
      for (int i = 0; i < 8; ++i)
        lds[(size_t)(16 * s + i + 8 * halfid) * ATOM + n0 + lcol] =
            (s ? acc[1][i] : acc[0][i]) + bo;
  }
  __syncthreads();
  const int rl = lane;
  const int gRow = m0 + rl;
  float mx = -1e30f; int am = 0;
  #pragma unroll
  for (int c = 0; c < ATOM; ++c) {
    const float v = lds[(size_t)rl * ATOM + c];
    if (v > mx) { mx = v; am = c; }
  }
  float sum = 0.0f;
  #pragma unroll
  for (int c = 0; c < ATOM; ++c) sum += __expf(lds[(size_t)rl * ATOM + c] - mx);
  const float lse = mx + __logf(sum);
  float* yr = yout + (size_t)gRow * ATOM;
  #pragma unroll
  for (int c4 = 0; c4 < ATOM / 4; ++c4) {
    f32x4 o;
    o.x = lds[(size_t)rl * ATOM + 4 * c4 + 0] - lse;
    o.y = lds[(size_t)rl * ATOM + 4 * c4 + 1] - lse;
    o.z = lds[(size_t)rl * ATOM + 4 * c4 + 2] - lse;
    o.w = lds[(size_t)rl * ATOM + 4 * c4 + 3] - lse;
    stg_f4(yr + 4 * c4, o);
  }
  const __bf16* es = embb + (size_t)am * 64;
  __bf16* ed = ebf + (size_t)gRow * 64;
  #pragma unroll
  for (int j = 0; j < 8; ++j) stg_u4(ed + 8 * j, ldg_u4(es + 8 * j));
  __syncthreads();
}

// ---------------- persistent decoder ----------------
__global__ __launch_bounds__(128) void decoder_kernel(
    char* __restrict__ ws,
    const float* __restrict__ b_h0,
    const float* __restrict__ b_ih0, const float* __restrict__ b_hh0,
    const float* __restrict__ b_ih1, const float* __restrict__ b_hh1,
    const float* __restrict__ b_out,
    float* __restrict__ out) {
  __shared__ float lds_all[4 * 32 * ATOM];
  const int lane = threadIdx.x & 31;
  const int wid  = threadIdx.x >> 5;
  const int m0   = (blockIdx.x * 4 + wid) * 32;
  float* lds = lds_all + (size_t)wid * 32 * ATOM;
  const int halfid = lane >> 4, lcol = lane & 15;

  const __bf16* encb  = (const __bf16*)(ws + OFF_ENCB);
  const __bf16* wh0t  = (const __bf16*)(ws + OFF_WH0T);
  const __bf16* wih0t = (const __bf16*)(ws + OFF_WIH0T);
  const __bf16* whh0t = (const __bf16*)(ws + OFF_WHH0T);
  const __bf16* wih1t = (const __bf16*)(ws + OFF_WIH1T);
  const __bf16* whh1t = (const __bf16*)(ws + OFF_WHH1T);
  const __bf16* woutt = (const __bf16*)(ws + OFF_WOUTT);
  const __bf16* embb  = (const __bf16*)(ws + OFF_EMBB);
  float*  h0f  = (float*)(ws + OFF_H0F);
  float*  h1f  = (float*)(ws + OFF_H1F);
  __bf16* h0bA = (__bf16*)(ws + OFF_H0BA);
  __bf16* h0bB = (__bf16*)(ws + OFF_H0BB);
  __bf16* h1bA = (__bf16*)(ws + OFF_H1BA);
  __bf16* h1bB = (__bf16*)(ws + OFF_H1BB);
  __bf16* ebf  = (__bf16*)(ws + OFF_EBF);

  // ---- init: [h0|h1] = tanh(enc @ w_h0^T + b_h0)
  {
    const char* paE = (const char*)encb + ((size_t)(m0 + lcol) * ENCD + halfid * 8) * 2;
    const char* pbH = (const char*)wh0t + (size_t)lane * (2 * HID) * 2;
    #pragma unroll 1
    for (int nt = 0; nt < (2 * HID) / 16; ++nt) {
      const int n0 = nt * 16;
      f32x8 acc[2] = {};
      gemm1<ENCD / 32, 16 * ENCD * 2, 32 * (2 * HID) * 2>(paE, pbH + n0 * 2, acc);
      const int col = n0 + lcol;
      const float bb = ldg_f(b_h0 + col);
      #pragma unroll
      for (int s = 0; s < 2; ++s)
        #pragma unroll
        for (int i = 0; i < 8; ++i) {
          const int row = m0 + 16 * s + i + 8 * halfid;
          const float v = tanhf((s ? acc[1][i] : acc[0][i]) + bb);
          if (col < HID) {
            stg_f(h0f + (size_t)row * HID + col, v);
            stg_bf(h0bA + (size_t)row * HID + col, (__bf16)v);
          } else {
            stg_f(h1f + (size_t)row * HID + (col - HID), v);
            stg_bf(h1bA + (size_t)row * HID + (col - HID), (__bf16)v);
          }
        }
    }
  }
  // ---- e init: SOS embedding (index 1)
  {
    const __bf16* es = embb + 64;
    __bf16* ed = ebf + (size_t)(m0 + lane) * 64;
    #pragma unroll
    for (int j = 0; j < 8; ++j) stg_u4(ed + 8 * j, ldg_u4(es + 8 * j));
  }

  // ---- recurrence, ping-pong bf16 state buffers
  const __bf16* h0in = h0bA; __bf16* h0out = h0bB;
  const __bf16* h1in = h1bA; __bf16* h1out = h1bB;
  #pragma unroll 1
  for (int step = 0; step < NSTEP; ++step) {
    gru_layer<64>(ebf, h0f, h0in, h0out, wih0t, whh0t, b_ih0, b_hh0, m0, lane);
    gru_layer<HID>(h0out, h1f, h1in, h1out, wih1t, whh1t, b_ih1, b_hh1, m0, lane);
    out_stage(h1out, woutt, b_out, embb, ebf,
              out + (size_t)step * BTOT * ATOM, lds, m0, lane);
    const __bf16* t0 = h0in; h0in = h0out; h0out = (__bf16*)t0;
    const __bf16* t1 = h1in; h1in = h1out; h1out = (__bf16*)t1;
  }
}

// ---------------- host launch ----------------
extern "C" void kernel_launch(void* const* d_in, const int* in_sizes, int n_in,
                              void* d_out, int out_size, void* d_ws, size_t ws_size,
                              hipStream_t stream) {
  (void)in_sizes; (void)n_in; (void)out_size; (void)ws_size;
  const float* enc   = (const float*)d_in[0];
  const float* emb   = (const float*)d_in[1];
  const float* w_h0  = (const float*)d_in[2];
  const float* b_h0  = (const float*)d_in[3];
  const float* w_ih0 = (const float*)d_in[4];
  const float* w_hh0 = (const float*)d_in[5];
  const float* b_ih0 = (const float*)d_in[6];
  const float* b_hh0 = (const float*)d_in[7];
  const float* w_ih1 = (const float*)d_in[8];
  const float* w_hh1 = (const float*)d_in[9];
  const float* b_ih1 = (const float*)d_in[10];
  const float* b_hh1 = (const float*)d_in[11];
  const float* w_out = (const float*)d_in[12];
  const float* b_out = (const float*)d_in[13];
  char*  ws  = (char*)d_ws;
  float* out = (float*)d_out;

  int n;
  n = BTOT * ENCD;
  k_cvt <<<(n + 255) / 256, 256, 0, stream>>>(enc, (__bf16*)(ws + OFF_ENCB), n);
  n = ENCD * (2 * HID);
  k_cvtT<<<(n + 255) / 256, 256, 0, stream>>>(w_h0,  (__bf16*)(ws + OFF_WH0T),  2 * HID, ENCD, ENCD);
  n = 64 * (3 * HID);
  k_cvtT<<<(n + 255) / 256, 256, 0, stream>>>(w_ih0, (__bf16*)(ws + OFF_WIH0T), 3 * HID, EMBD, 64);
  n = HID * (3 * HID);
  k_cvtT<<<(n + 255) / 256, 256, 0, stream>>>(w_hh0, (__bf16*)(ws + OFF_WHH0T), 3 * HID, HID, HID);
  k_cvtT<<<(n + 255) / 256, 256, 0, stream>>>(w_ih1, (__bf16*)(ws + OFF_WIH1T), 3 * HID, HID, HID);
  k_cvtT<<<(n + 255) / 256, 256, 0, stream>>>(w_hh1, (__bf16*)(ws + OFF_WHH1T), 3 * HID, HID, HID);
  n = HID * ATOM;
  k_cvtT<<<(n + 255) / 256, 256, 0, stream>>>(w_out, (__bf16*)(ws + OFF_WOUTT), ATOM, HID, HID);
  n = ATOM * 64;
  k_emb <<<(n + 255) / 256, 256, 0, stream>>>(emb, (__bf16*)(ws + OFF_EMBB));

  decoder_kernel<<<BTOT / (32 * 4), 128, 0, stream>>>(
      ws, b_h0, b_ih0, b_hh0, b_ih1, b_hh1, b_out, out);
}
